// GATLayer_32152125178258
// MI455X (gfx1250) — matrix operand order repaired and checked by reading, not compiled, not run on hardware
//
#include <hip/hip_runtime.h>

#define ALPHA 0.2f
#define LOG2E 1.44269504088896340736f
#define B_ 4
#define N_ 2048
#define F_ 128
#define H_ 8
#define D_ 32
#define C_ 128   // H*D

typedef __attribute__((ext_vector_type(16))) _Float16 v16h;
typedef __attribute__((ext_vector_type(8)))  float    v8f;

static __device__ __forceinline__ float fast_exp2(float x) {
#if __has_builtin(__builtin_amdgcn_exp2f)
    return __builtin_amdgcn_exp2f(x);        // raw v_exp_f32
#else
    return exp2f(x);
#endif
}

// ---------------------------------------------------------------------------
// Kernel 1: Wh = x@W (fp32), emit WhT as f16 [B][C][N] for WMMA B-operands,
// and s[b,n,h] = sum_d Wh[b,n,h,d]*a[h,d].
// One block = 16 rows of one batch; 128 threads (one per output channel).
// ---------------------------------------------------------------------------
__global__ __launch_bounds__(128) void gat_proj(const float* __restrict__ x,
                                                const float* __restrict__ W,
                                                const float* __restrict__ a,
                                                _Float16* __restrict__ WhT,
                                                float* __restrict__ s)
{
    __shared__ float xs[16][F_];
    const int t  = threadIdx.x;
    const int b  = blockIdx.x / (N_ / 16);
    const int n0 = (blockIdx.x % (N_ / 16)) * 16;

    const float* xrow = x + (size_t)(b * N_ + n0) * F_;
    for (int r = 0; r < 16; ++r) xs[r][t] = xrow[r * F_ + t];
    __syncthreads();

    float acc[16];
#pragma unroll
    for (int r = 0; r < 16; ++r) acc[r] = 0.f;
    for (int k = 0; k < F_; ++k) {
        const float wv = W[k * C_ + t];           // broadcast-friendly L2 reads
#pragma unroll
        for (int r = 0; r < 16; ++r) acc[r] = fmaf(xs[r][k], wv, acc[r]);
    }

    const float av = a[t];                        // a flattened [H*D] == channel t
#pragma unroll
    for (int r = 0; r < 16; ++r)
        WhT[(size_t)(b * C_ + t) * N_ + n0 + r] = (_Float16)acc[r];

    __syncthreads();                              // xs reads done; reuse as products
    for (int r = 0; r < 16; ++r) xs[r][t] = acc[r] * av;
    __syncthreads();
    {   // 128 threads == 16 rows x 8 heads
        const int r = t >> 3, h = t & 7;
        float ss = 0.f;
        for (int d = 0; d < D_; ++d) ss += xs[r][h * D_ + d];
        s[(size_t)(b * N_ + n0 + r) * H_ + h] = ss;
    }
}

// ---------------------------------------------------------------------------
// Kernel 2: M[b,h] = max_n s[b,n,h]  (softmax stabilizer upper bound)
// ---------------------------------------------------------------------------
__global__ __launch_bounds__(256) void gat_rowmax(const float* __restrict__ s,
                                                  float* __restrict__ M)
{
    __shared__ float red[256];
    const int b = blockIdx.x / H_, h = blockIdx.x % H_;
    float m = -1e30f;
    for (int n = threadIdx.x; n < N_; n += 256)
        m = fmaxf(m, s[(size_t)(b * N_ + n) * H_ + h]);
    red[threadIdx.x] = m;
    __syncthreads();
    for (int off = 128; off > 0; off >>= 1) {
        if (threadIdx.x < off)
            red[threadIdx.x] = fmaxf(red[threadIdx.x], red[threadIdx.x + off]);
        __syncthreads();
    }
    if (threadIdx.x == 0) M[blockIdx.x] = red[0];
}

// ---------------------------------------------------------------------------
// Kernel 3: fused masked-softmax aggregation.
// Block = one (b, i-tile of 16); 8 waves, wave w == head w.
// adj tile read ONCE per block (non-temporal), ballot-packed to 16 bitmasks
// in LDS, consumed by all 8 heads. Manually unrolled x2 with ping-pong LDS
// buffers and two fixed B-register sets (no rotation copies). Scores in the
// log2e-scaled domain: arg = max(sv+c1, ALPHA*sv+c2); p = v_exp_f32(arg).
// ---------------------------------------------------------------------------
__global__ __launch_bounds__(256) void gat_attn(const int* __restrict__ adj,
                                                const float* __restrict__ s,
                                                const float* __restrict__ M,
                                                const _Float16* __restrict__ WhT,
                                                float* __restrict__ out)
{
    __shared__ __align__(16) float sjl[2][H_][32];   // transposed: [head][j], *log2e
    __shared__ unsigned adjb[2][16];

    const int t   = threadIdx.x;
    const int w   = t >> 5;          // wave index == head
    const int l   = t & 31;          // lane within wave32
    const int h   = w;
    const int b   = blockIdx.y;
    const int i0  = blockIdx.x * 16;
    const int row = l & 15;          // A-matrix row / B-matrix column
    const int g   = l >> 4;          // half-wave select

    // log2e-scaled domain: leaky(x)*c == leaky(x*c) for c>0, max commutes too
    const float si = s[(size_t)(b * N_ + i0 + row) * H_ + h] * LOG2E;
    const float mb = M[b * H_ + h] * LOG2E;
    float mstab = si + mb;
    mstab = fmaxf(mstab, ALPHA * mstab);            // scaled upper bound
    const float c1 = si - mstab;                    // arg = max(sv+c1, A*sv+c2) <= 0
    const float c2 = fmaf(ALPHA, si, -mstab);

    v8f acc0 = {};       // D columns 0..15 of this head
    v8f acc1 = {};       // D columns 16..31
    float den0 = 0.f, den1 = 0.f, den2 = 0.f, den3 = 0.f;

    const size_t    adjbase = ((size_t)b * N_ + i0) * N_;
    const _Float16* whb     = WhT + (size_t)(b * C_ + h * D_) * N_;
    const _Float16* p0      = whb + (size_t)(0 * 16 + row) * N_ + g * 16;
    const _Float16* p1      = whb + (size_t)(1 * 16 + row) * N_ + g * 16;

    // cooperative tile loader: s_j (coalesced, transposed into LDS) + adj ballot
    auto load_tile = [&](int j0, int buf) {
        const float sv = s[(size_t)(b * N_ + j0) * H_ + t] * LOG2E;  // addr = j0*8 + t
        sjl[buf][t & 7][t >> 3] = sv;
        const int r0 = w * 2;
        const int v0 = __builtin_nontemporal_load(&adj[adjbase + (size_t)r0 * N_ + j0 + l]);
        const unsigned m0 = (unsigned)__ballot(v0 != 0);
        const int v1 = __builtin_nontemporal_load(&adj[adjbase + (size_t)(r0 + 1) * N_ + j0 + l]);
        const unsigned m1 = (unsigned)__ballot(v1 != 0);
        if (l == 0) { adjb[buf][r0] = m0; adjb[buf][r0 + 1] = m1; }
    };

    // per-tile compute: build P (A-operand) and run both WMMAs
    auto compute_tile = [&](int buf, const v16h& bb0, const v16h& bb1) {
        const unsigned amask = adjb[buf][row];

        // lane's 16 s_j values: one contiguous 16-float run -> 4x ds_load_b128
        float sv[16];
        {
            const float4* sp = (const float4*)&sjl[buf][h][0];
            const float4 q0 = sp[g * 4 + 0];
            const float4 q1 = sp[g * 4 + 1];
            const float4 q2 = sp[g * 4 + 2];
            const float4 q3 = sp[g * 4 + 3];
            sv[0] = q0.x;  sv[1] = q0.y;  sv[2]  = q0.z;  sv[3]  = q0.w;
            sv[4] = q1.x;  sv[5] = q1.y;  sv[6]  = q1.z;  sv[7]  = q1.w;
            sv[8] = q2.x;  sv[9] = q2.y;  sv[10] = q2.z;  sv[11] = q2.w;
            sv[12] = q3.x; sv[13] = q3.y; sv[14] = q3.z;  sv[15] = q3.w;
        }

        v16h ap;
#pragma unroll
        for (int e = 0; e < 16; ++e) {
            const int jj = e + g * 16;                          // K index
            const float arg = fmaxf(sv[e] + c1, fmaf(ALPHA, sv[e], c2));
            float p = fast_exp2(arg);                           // v_exp_f32, arg <= 0
            p = ((amask >> jj) & 1u) ? p : 0.f;                 // v_cndmask
            if      ((e & 3) == 0) den0 += p;
            else if ((e & 3) == 1) den1 += p;
            else if ((e & 3) == 2) den2 += p;
            else                   den3 += p;
            ap[e] = (_Float16)p;
        }

        acc0 = __builtin_amdgcn_wmma_f32_16x16x32_f16(false, ap, false, bb0,
                                                      (short)0, acc0, false, false);
        acc1 = __builtin_amdgcn_wmma_f32_16x16x32_f16(false, ap, false, bb1,
                                                      (short)0, acc1, false, false);
    };

    // prologue: tile 0 into buffer 0 / register set A
    load_tile(0, 0);
    v16h bA0 = *(const v16h*)(p0);
    v16h bA1 = *(const v16h*)(p1);

    const int NT = N_ / 32;                        // 64, even
    for (int T = 0; T < NT; T += 2) {
        const int j0 = T * 32;

        // ---- phase A: compute tile T (buf 0, set A); stage tile T+1 ----
        __syncthreads();                           // buf 0 resident
        load_tile(j0 + 32, 1);
        const v16h bB0 = *(const v16h*)(p0 + j0 + 32);
        const v16h bB1 = *(const v16h*)(p1 + j0 + 32);
        __builtin_prefetch(&adj[adjbase + (size_t)(w * 2) * N_ + j0 + 64 + l], 0, 0);
        compute_tile(0, bA0, bA1);

        // ---- phase B: compute tile T+1 (buf 1, set B); stage tile T+2 ----
        __syncthreads();                           // buf 1 resident
        if (T + 2 < NT) {
            load_tile(j0 + 64, 0);
            bA0 = *(const v16h*)(p0 + j0 + 64);
            bA1 = *(const v16h*)(p1 + j0 + 64);
            __builtin_prefetch(&adj[adjbase + (size_t)(w * 2) * N_ + j0 + 96 + l], 0, 0);
        }
        compute_tile(1, bB0, bB1);
    }

    // lanes l and l^16 hold complementary j-halves of the same row
    float denom = (den0 + den1) + (den2 + den3);
    denom += __shfl_xor(denom, 16, 32);
    const float inv = denom > 0.f ? 1.f / denom : 0.f;

    float invr[8];
#pragma unroll
    for (int r = 0; r < 8; ++r) invr[r] = __shfl(inv, r + 8 * g, 32);  // lane m holds row m

    float* outb = out + (size_t)(b * N_ + i0) * C_ + h * D_;
#pragma unroll
    for (int r = 0; r < 8; ++r) {
        const int m = r + 8 * g;                          // C-layout row
        outb[(size_t)m * C_ + 0 * 16 + row] = acc0[r] * invr[r];
        outb[(size_t)m * C_ + 1 * 16 + row] = acc1[r] * invr[r];
    }
}

// ---------------------------------------------------------------------------
extern "C" void kernel_launch(void* const* d_in, const int* in_sizes, int n_in,
                              void* d_out, int out_size, void* d_ws, size_t ws_size,
                              hipStream_t stream)
{
    const float* x   = (const float*)d_in[0];   // [B,N,F]
    const int*   adj = (const int*)  d_in[1];   // [B,N,N]
    const float* W   = (const float*)d_in[2];   // [F,H*D]
    const float* a   = (const float*)d_in[3];   // [H,D]
    float*       out = (float*)d_out;           // [B,N,H*D]

    char* ws = (char*)d_ws;
    _Float16* WhT = (_Float16*)ws;                                        // 2 MB
    float*    s   = (float*)(ws + (size_t)B_ * C_ * N_ * 2);              // 256 KB
    float*    M   = (float*)(ws + (size_t)B_ * C_ * N_ * 2
                                + (size_t)B_ * N_ * H_ * 4);              // 128 B

    gat_proj  <<<dim3(B_ * N_ / 16), dim3(128), 0, stream>>>(x, W, a, WhT, s);
    gat_rowmax<<<dim3(B_ * H_),      dim3(256), 0, stream>>>(s, M);
    gat_attn  <<<dim3(N_ / 16, B_),  dim3(256), 0, stream>>>(adj, s, M, WhT, out);
}
